// XlPosition_55233279427082
// MI455X (gfx1250) — compile-verified, matrix operand order checked by reading, not run on hardware
//
#include <hip/hip_runtime.h>

#define QLEN   1024
#define KLEN   2048
#define NH     16
#define BS     4
#define DH     64
#define DMODEL 1024

typedef __bf16 bf16_t;
typedef __attribute__((ext_vector_type(16))) __bf16 v16bf;
typedef __attribute__((ext_vector_type(8)))  __bf16 v8bf;
typedef __attribute__((ext_vector_type(8)))  float   v8f;

static __device__ __forceinline__ v16bf cat8(v8bf lo, v8bf hi) {
  return __builtin_shufflevector(lo, hi, 0,1,2,3,4,5,6,7,8,9,10,11,12,13,14,15);
}

// ---------------------------------------------------------------------------
// fp32 -> bf16 conversion (grid-stride)
// ---------------------------------------------------------------------------
__global__ __launch_bounds__(256) void f32_to_bf16_kernel(
    const float* __restrict__ in, bf16_t* __restrict__ out, int n) {
  int idx = blockIdx.x * blockDim.x + threadIdx.x;
  int stride = gridDim.x * blockDim.x;
  for (int i = idx; i < n; i += stride) out[i] = (bf16_t)in[i];
}

// ---------------------------------------------------------------------------
// r_head_k = r @ W_r^T   (M=2048 j, N=1024 nd, K=1024), bf16 WMMA, f32 accum.
// A = r_bf16 [j][m] (K-contiguous), B = W_r_bf16 [nd][m] (K-contiguous).
// Emits both fp32 (for D) and bf16 (for the B GEMM) copies.
// One wave per 16x16 tile; 8 waves/block.
// ---------------------------------------------------------------------------
__global__ __launch_bounds__(256) void rheadk_gemm_kernel(
    const bf16_t* __restrict__ Ab, const bf16_t* __restrict__ Bb,
    float* __restrict__ rk_f32, bf16_t* __restrict__ rk_b16) {
  const int tid  = threadIdx.x;
  const int wave = tid >> 5;
  const int lane = tid & 31;
  const int half = lane >> 4;   // which half of the wave (K-chunk select)
  const int l16  = lane & 15;   // row (A) / col (B) within tile

  const int j0 = blockIdx.x * 16;
  const int n0 = (blockIdx.y * 8 + wave) * 16;

  const bf16_t* arow = Ab + (size_t)(j0 + l16) * DMODEL;
  const bf16_t* brow = Bb + (size_t)(n0 + l16) * DMODEL;

  v8f acc = {};
  for (int kk = 0; kk < DMODEL; kk += 32) {
    const v8bf* pa = (const v8bf*)(arow + kk + half * 8);
    v16bf afrag = cat8(pa[0], pa[2]);   // K = half*8..+8 and K+16..+8
    const v8bf* pb = (const v8bf*)(brow + kk + half * 8);
    v16bf bfrag = cat8(pb[0], pb[2]);
    acc = __builtin_amdgcn_wmma_f32_16x16x32_bf16(
        false, afrag, false, bfrag, (short)0, acc, false, false);
  }

#pragma unroll
  for (int v = 0; v < 8; ++v) {
    const int row = j0 + v + half * 8;   // M = v (+8 for upper half-wave)
    const int col = n0 + l16;            // N = lane%16
    const float x = acc[v];
    rk_f32[(size_t)row * DMODEL + col] = x;
    rk_b16[(size_t)row * DMODEL + col] = (bf16_t)x;
  }
}

// ---------------------------------------------------------------------------
// C[j][n] = sum_{b,d} r_w_bias[n][d] * w_head_k[j][b][n][d]
// D[j][n] = sum_{d}   r_r_bias[n][d] * r_head_k[j][n][d]
// ---------------------------------------------------------------------------
__global__ __launch_bounds__(256) void bias_kernel(
    const float* __restrict__ whk, const float* __restrict__ rk_f32,
    const float* __restrict__ rwb, const float* __restrict__ rrb,
    float* __restrict__ Cb, float* __restrict__ Db) {
  const int tid = threadIdx.x;
  const int n  = tid & 15;
  const int jl = tid >> 4;
  const int j  = blockIdx.x * 16 + jl;

  float c = 0.f, d = 0.f;
  const float4* wv4 = (const float4*)(rwb + n * DH);
  const float4* rv4 = (const float4*)(rrb + n * DH);
  const float4* kv4 = (const float4*)(rk_f32 + (size_t)j * DMODEL + n * DH);
#pragma unroll 4
  for (int d4 = 0; d4 < DH / 4; ++d4) {
    float4 wv = wv4[d4];
    float4 rv = rv4[d4];
    float4 kv = kv4[d4];
    d += rv.x * kv.x + rv.y * kv.y + rv.z * kv.z + rv.w * kv.w;
#pragma unroll
    for (int b = 0; b < BS; ++b) {
      float4 hv = ((const float4*)(whk + (((size_t)j * BS + b) * NH + n) * DH))[d4];
      c += wv.x * hv.x + wv.y * hv.y + wv.z * hv.z + wv.w * hv.w;
    }
  }
  Cb[j * NH + n] = c;
  Db[j * NH + n] = d;
}

// ---------------------------------------------------------------------------
// Zero-pad diagonal of the rel-shift: out[i][i+1025][b][n] = C[i+1025][n]
// (BD contribution is exactly 0 there), for i in [0, 1023).
// ---------------------------------------------------------------------------
__global__ __launch_bounds__(256) void diag_kernel(
    const float* __restrict__ Cb, float* __restrict__ out) {
  int idx = blockIdx.x * blockDim.x + threadIdx.x;
  if (idx >= (QLEN - 1) * BS * NH) return;
  const int i  = idx >> 6;
  const int bn = idx & 63;
  const int b  = bn >> 4;
  const int n  = bn & 15;
  const int j  = i + 1025;
  out[(((size_t)i * KLEN + j) * BS + b) * NH + n] = Cb[j * NH + n];
}

// ---------------------------------------------------------------------------
// Main fused kernel: per workgroup, fixed batch b, B-tile i0..i0+16 x k0..k0+16
// for ALL 16 heads via bf16 WMMA.
//  1) Async-stage the contiguous 32KB Rk tile (16 k x 16 n x 64 d bf16) into
//     LDS with global_load_async_to_lds_b128 (ASYNCcnt path), overlapping the
//     per-wave Q fragment global loads with the copies.
//  2) WMMA with B-operand fragments served from LDS (ds_load_b128).
//  3) Stage fp32 results in LDS [i][k][n], then cooperative rel-shift scatter
//     epilogue with n-contiguous (64B-line) stores:
//       B[i][k]: k >= 1023-i  -> out[i  ][k+i-1023]
//                k <  1023-i  -> out[i-1][k+i+1025]   (dropped when i==0)
//       out value = B + D[k][n] + C[j][n]
// ---------------------------------------------------------------------------
__global__ __launch_bounds__(256) void b_gemm_scatter_kernel(
    const bf16_t* __restrict__ qb,   // [QLEN][BS][NH][DH] bf16
    const bf16_t* __restrict__ rkb,  // [KLEN][NH][DH] bf16
    const float* __restrict__ Cb,    // [KLEN][NH]
    const float* __restrict__ Db,    // [KLEN][NH]
    float* __restrict__ out) {       // [QLEN][KLEN][BS][NH]
  __shared__ __align__(16) bf16_t rk_tile[16 * NH * DH];  // 32 KB [kloc][n][d]
  __shared__ float acc_lds[16 * 16 * 17];                 // ~17 KB, n padded

  const int k0 = blockIdx.x * 16;
  const int i0 = blockIdx.y * 16;
  const int b  = blockIdx.z;

  const int tid  = threadIdx.x;
  const int wave = tid >> 5;
  const int lane = tid & 31;
  const int half = lane >> 4;
  const int l16  = lane & 15;

  // ---- 1) issue async global->LDS copies of the Rk tile (contiguous 32KB) --
  {
    const char* src = (const char*)(rkb + (size_t)k0 * NH * DH);
    const unsigned lbase = (unsigned)(uintptr_t)(&rk_tile[0]);
#pragma unroll
    for (int t = 0; t < 8; ++t) {
      const unsigned byte_off = (unsigned)tid * 16u + (unsigned)t * 4096u;
      unsigned long long gaddr = (unsigned long long)(uintptr_t)(src + byte_off);
      unsigned loff = lbase + byte_off;
      asm volatile("global_load_async_to_lds_b128 %0, %1, off"
                   :: "v"(loff), "v"(gaddr) : "memory");
    }
  }

  // ---- overlap: per-wave Q (A-operand) fragments from global -------------
  // wave handles heads {wave, wave+8}; A rows i0..i0+15, K = d (2 x 32)
  v16bf afrag[2][2];
#pragma unroll
  for (int nsel = 0; nsel < 2; ++nsel) {
    const int n = wave + nsel * 8;
    const bf16_t* qrow = qb + (((size_t)(i0 + l16) * BS + b) * NH + n) * DH;
#pragma unroll
    for (int kk = 0; kk < 2; ++kk) {
      const v8bf* pa = (const v8bf*)(qrow + kk * 32 + half * 8);
      afrag[nsel][kk] = cat8(pa[0], pa[2]);
    }
  }

  asm volatile("s_wait_asynccnt 0x0" ::: "memory");
  __syncthreads();

  // ---- 2) compute: B-operand fragments from LDS --------------------------
#pragma unroll
  for (int nsel = 0; nsel < 2; ++nsel) {
    const int n = wave + nsel * 8;
    // B operand: cols k = k0 + l16 -> LDS row l16; K = d chunks
    const bf16_t* rrow = rk_tile + ((l16 * NH + n) * DH);
    v8f acc = {};
#pragma unroll
    for (int kk = 0; kk < 2; ++kk) {
      const v8bf* pbl = (const v8bf*)(rrow + kk * 32 + half * 8);
      v16bf bfrag = cat8(pbl[0], pbl[2]);
      acc = __builtin_amdgcn_wmma_f32_16x16x32_bf16(
          false, afrag[nsel][kk], false, bfrag, (short)0, acc, false, false);
    }
#pragma unroll
    for (int v = 0; v < 8; ++v) {
      const int iloc = v + half * 8;
      const int kloc = l16;
      acc_lds[(iloc * 16 + kloc) * 17 + n] = acc[v];
    }
  }
  __syncthreads();

  // ---- 3) epilogue: rel-shift scatter, n-contiguous stores ---------------
  const int n = tid & 15;
  for (int p = tid >> 4; p < 16 * 16; p += 16) {
    const int iloc = p >> 4;
    const int kloc = p & 15;
    const int i = i0 + iloc;
    const int k = k0 + kloc;
    int orow, j;
    if (k >= 1023 - i) { orow = i;     j = k + i - 1023; }
    else               { orow = i - 1; j = k + i + 1025; }
    if (orow < 0) continue;  // dropped first row of the rel-shift view
    const float val = acc_lds[p * 17 + n] + Db[k * NH + n] + Cb[j * NH + n];
    out[(((size_t)orow * KLEN + j) * BS + b) * NH + n] = val;
  }
}

// ---------------------------------------------------------------------------
extern "C" void kernel_launch(void* const* d_in, const int* in_sizes, int n_in,
                              void* d_out, int out_size, void* d_ws, size_t ws_size,
                              hipStream_t stream) {
  (void)in_sizes; (void)n_in; (void)out_size; (void)ws_size;
  const float* w_head_q = (const float*)d_in[0];
  const float* w_head_k = (const float*)d_in[1];
  const float* r        = (const float*)d_in[2];
  const float* W_r      = (const float*)d_in[3];
  const float* r_w_bias = (const float*)d_in[4];
  const float* r_r_bias = (const float*)d_in[5];
  float* out = (float*)d_out;

  // workspace layout (bytes, all 256-aligned)
  char* ws = (char*)d_ws;
  bf16_t* qb  = (bf16_t*)(ws + 0);          //  8 MB  Q bf16
  bf16_t* rb  = (bf16_t*)(ws + 8388608);    //  4 MB  r bf16
  bf16_t* wrb = (bf16_t*)(ws + 12582912);   //  2 MB  W_r bf16
  float*  rkf = (float*) (ws + 14680064);   //  8 MB  r_head_k fp32
  bf16_t* rkb = (bf16_t*)(ws + 23068672);   //  4 MB  r_head_k bf16
  float*  Cb  = (float*) (ws + 27262976);   // 128 KB C[j][n]
  float*  Db  = (float*) (ws + 27394048);   // 128 KB D[j][n]

  f32_to_bf16_kernel<<<2048, 256, 0, stream>>>(w_head_q, qb, QLEN * BS * NH * DH);
  f32_to_bf16_kernel<<<1024, 256, 0, stream>>>(r, rb, KLEN * DMODEL);
  f32_to_bf16_kernel<<<512,  256, 0, stream>>>(W_r, wrb, NH * DH * DMODEL);

  rheadk_gemm_kernel<<<dim3(KLEN / 16, 8), 256, 0, stream>>>(rb, wrb, rkf, rkb);

  bias_kernel<<<KLEN / 16, 256, 0, stream>>>(w_head_k, rkf, r_w_bias, r_r_bias, Cb, Db);

  diag_kernel<<<((QLEN - 1) * BS * NH + 255) / 256, 256, 0, stream>>>(Cb, out);

  b_gemm_scatter_kernel<<<dim3(KLEN / 16, QLEN / 16, BS), 256, 0, stream>>>(
      qb, rkb, Cb, Db, out);
}
